// GNN_33097017983485
// MI455X (gfx1250) — matrix operand order repaired and checked by reading, not compiled, not run on hardware
//
#include <hip/hip_runtime.h>
#include <hip/hip_bf16.h>

typedef __attribute__((ext_vector_type(16))) _Float16 v16h;
typedef __attribute__((ext_vector_type(8)))  _Float16 v8h;
typedef __attribute__((ext_vector_type(8)))  float    v8f;

#define H 64
#define NG 512

__device__ __forceinline__ v8f wmma_f16(v16h a, v16h b, v8f c) {
  return __builtin_amdgcn_wmma_f32_16x16x32_f16(false, a, false, b, (short)0, c,
                                                false, false);
}

// ---------------------------------------------------------------- elementwise
__global__ void k_f32_to_f16(const float* __restrict__ src,
                             _Float16* __restrict__ dst, long n) {
  long i = (long)blockIdx.x * blockDim.x + threadIdx.x;
  long s = (long)gridDim.x * blockDim.x;
  for (; i < n; i += s) dst[i] = (_Float16)src[i];
}

// W [K][64] f32 -> Wt [64][Kpad] f16 (transposed, zero-padded along K)
__global__ void k_prep_wT(const float* __restrict__ W, _Float16* __restrict__ Wt,
                          int K, int Kpad) {
  int i = blockIdx.x * blockDim.x + threadIdx.x;
  int total = 64 * Kpad;
  if (i >= total) return;
  int n = i / Kpad, k = i % Kpad;
  Wt[i] = (k < K) ? (_Float16)W[k * 64 + n] : (_Float16)0.0f;
}

// ---------------------------------------------------------------- GEMM K=64
// A [M][64] f16 row-major, Wt [64][64] f16 (row = out col, contiguous K).
// One wave per 16-row tile, 64 output cols (4 WMMA acc), K-loop of 2.
__global__ void __launch_bounds__(256)
k_gemm64(const _Float16* __restrict__ A, const _Float16* __restrict__ Wt,
         const float* __restrict__ bias, float* __restrict__ Z, long Mtiles) {
  long tile = (long)blockIdx.x * 8 + (threadIdx.x >> 5);
  if (tile >= Mtiles) return;
  unsigned lane = threadIdx.x & 31u;
  unsigned g = lane >> 4, r = lane & 15u;
  const _Float16* Abase = A + tile * 16 * 64;
  v8f c0 = {}, c1 = {}, c2 = {}, c3 = {};
#pragma unroll
  for (int kb = 0; kb < 64; kb += 32) {
    const _Float16* ap = Abase + (size_t)r * 64 + kb + g * 16;
    v8h alo = *(const v8h*)ap;
    v8h ahi = *(const v8h*)(ap + 8);
    v16h a;
#pragma unroll
    for (int i = 0; i < 8; ++i) { a[i] = alo[i]; a[i + 8] = ahi[i]; }
    v16h b0 = *(const v16h*)(Wt + (0 * 16 + r) * 64 + kb + g * 16);
    v16h b1 = *(const v16h*)(Wt + (1 * 16 + r) * 64 + kb + g * 16);
    v16h b2 = *(const v16h*)(Wt + (2 * 16 + r) * 64 + kb + g * 16);
    v16h b3 = *(const v16h*)(Wt + (3 * 16 + r) * 64 + kb + g * 16);
    c0 = wmma_f16(a, b0, c0);
    c1 = wmma_f16(a, b1, c1);
    c2 = wmma_f16(a, b2, c2);
    c3 = wmma_f16(a, b3, c3);
  }
  v8f cc[4] = {c0, c1, c2, c3};
#pragma unroll
  for (int nt = 0; nt < 4; ++nt) {
    float bv = bias ? bias[nt * 16 + r] : 0.0f;
#pragma unroll
    for (int v = 0; v < 8; ++v) {
      long row = tile * 16 + g * 8 + v;
      Z[row * 64 + nt * 16 + r] = cc[nt][v] + bv;
    }
  }
}

// ------------------------------------------------------ edge tile GEMM (K=16)
// ea16 [E][16] f16, WeT [64][32] f16 (zero-padded K 16..31). Bias added.
__device__ __forceinline__ void edge_tile_gemm(const _Float16* __restrict__ ea16,
                                               long tile,
                                               const _Float16* __restrict__ WeT,
                                               const float* __restrict__ be,
                                               unsigned lane, v8f c[4]) {
  unsigned g = lane >> 4, r = lane & 15u;
  v16h a = g ? v16h{} : *(const v16h*)(ea16 + (tile * 16 + r) * 16);
#pragma unroll
  for (int nt = 0; nt < 4; ++nt) {
    v16h b = *(const v16h*)(WeT + (nt * 16 + r) * 32 + g * 16);
    c[nt] = wmma_f16(a, b, c[nt]);
  }
#pragma unroll
  for (int nt = 0; nt < 4; ++nt) {
    float bv = be[nt * 16 + r];
#pragma unroll
    for (int v = 0; v < 8; ++v) c[nt][v] += bv;
  }
}

// pass 1: column sum/sumsq of edge linear output (for BatchNorm stats)
__global__ void __launch_bounds__(256)
k_edge_pass1(const _Float16* __restrict__ ea16, const _Float16* __restrict__ WeT,
             const float* __restrict__ be, float* __restrict__ stats, long Etiles) {
  __shared__ float lsum[64], lsq[64];
  if (threadIdx.x < 64) { lsum[threadIdx.x] = 0.f; lsq[threadIdx.x] = 0.f; }
  __syncthreads();
  long tile = (long)blockIdx.x * 8 + (threadIdx.x >> 5);
  unsigned lane = threadIdx.x & 31u;
  if (tile < Etiles) {
    v8f c[4] = {v8f{}, v8f{}, v8f{}, v8f{}};
    edge_tile_gemm(ea16, tile, WeT, be, lane, c);
    unsigned r = lane & 15u;
#pragma unroll
    for (int nt = 0; nt < 4; ++nt) {
      float s = 0.f, q = 0.f;
#pragma unroll
      for (int v = 0; v < 8; ++v) { float z = c[nt][v]; s += z; q += z * z; }
      atomicAdd(&lsum[nt * 16 + r], s);
      atomicAdd(&lsq[nt * 16 + r], q);
    }
  }
  __syncthreads();
  if (threadIdx.x < 64) {
    unsafeAtomicAdd(&stats[threadIdx.x], lsum[threadIdx.x]);
    unsafeAtomicAdd(&stats[64 + threadIdx.x], lsq[threadIdx.x]);
  }
}

// pass 2: recompute edge GEMM, BN+ReLU, msg = relu(h[src]+ea), scatter to aggr
__global__ void __launch_bounds__(256)
k_edge_pass2(const _Float16* __restrict__ ea16, const _Float16* __restrict__ WeT,
             const float* __restrict__ be, const float* __restrict__ ss,
             const int* __restrict__ src, const int* __restrict__ dst,
             const float* __restrict__ h, float* __restrict__ aggr, long Etiles) {
  __shared__ float ts[8][64][16];  // [wave][col][edge-in-tile]
  int w = threadIdx.x >> 5;
  unsigned lane = threadIdx.x & 31u;
  long tile = (long)blockIdx.x * 8 + w;
  bool active = tile < Etiles;
  if (active) {
    v8f c[4] = {v8f{}, v8f{}, v8f{}, v8f{}};
    edge_tile_gemm(ea16, tile, WeT, be, lane, c);
    unsigned g = lane >> 4, r = lane & 15u;
#pragma unroll
    for (int nt = 0; nt < 4; ++nt)
#pragma unroll
      for (int v = 0; v < 8; ++v)
        ts[w][nt * 16 + r][g * 8 + v] = c[nt][v];
  }
  __syncthreads();
  if (active) {
    int e = lane >> 1;
    int cb = (lane & 1) * 32;
    long eidx = tile * 16 + e;
    long s = (long)src[eidx];
    long d = (long)dst[eidx];
    const float* hrow = h + s * 64;
    float* arow = aggr + d * 64;
#pragma unroll 4
    for (int c = 0; c < 32; ++c) {
      int col = cb + c;
      float ea = fmaxf(ss[col] * ts[w][col][e] + ss[64 + col], 0.0f);
      float m = fmaxf(hrow[col] + ea, 0.0f);
      unsafeAtomicAdd(&arow[col], m);
    }
  }
}

// ------------------------------------------------------------ BN stats / apply
__global__ void __launch_bounds__(256)
k_col_stats(const float* __restrict__ Z, long M, float* __restrict__ stats) {
  __shared__ float lsum[64], lsq[64];
  if (threadIdx.x < 64) { lsum[threadIdx.x] = 0.f; lsq[threadIdx.x] = 0.f; }
  __syncthreads();
  int col = threadIdx.x & 63;
  long row = (long)blockIdx.x * 4 + (threadIdx.x >> 6);
  long stride = (long)gridDim.x * 4;
  float s = 0.f, q = 0.f;
  for (; row < M; row += stride) {
    float z = Z[row * 64 + col];
    s += z; q += z * z;
  }
  atomicAdd(&lsum[col], s);
  atomicAdd(&lsq[col], q);
  __syncthreads();
  if (threadIdx.x < 64) {
    unsafeAtomicAdd(&stats[threadIdx.x], lsum[threadIdx.x]);
    unsafeAtomicAdd(&stats[64 + threadIdx.x], lsq[threadIdx.x]);
  }
}

__global__ void k_bn_finalize(const float* __restrict__ stats, float invM,
                              const float* __restrict__ g,
                              const float* __restrict__ b,
                              float* __restrict__ ss) {
  int c = threadIdx.x;
  if (c >= 64) return;
  float mean = stats[c] * invM;
  float var = stats[64 + c] * invM - mean * mean;
  float rstd = rsqrtf(var + 1e-5f);
  float sc = g[c] * rstd;
  ss[c] = sc;
  ss[64 + c] = b[c] - mean * sc;
}

// h(out) = relu(scale*Z+shift) (+ old h if residual); also emit f16 copy
__global__ void k_bn_relu(const float* __restrict__ Z, const float* __restrict__ ss,
                          float* __restrict__ hout, _Float16* __restrict__ h16,
                          long n, int residual) {
  long i = (long)blockIdx.x * blockDim.x + threadIdx.x;
  long s = (long)gridDim.x * blockDim.x;
  for (; i < n; i += s) {
    int col = (int)(i & 63);
    float v = fmaxf(ss[col] * Z[i] + ss[64 + col], 0.0f);
    if (residual) v += hout[i];
    hout[i] = v;
    h16[i] = (_Float16)v;
  }
}

// t16 = f16((1+eps[l])*h + aggr)
__global__ void k_combine(const float* __restrict__ h, const float* __restrict__ aggr,
                          const float* __restrict__ eps, int l,
                          _Float16* __restrict__ t16, long n) {
  float e1 = 1.0f + eps[l];
  long i = (long)blockIdx.x * blockDim.x + threadIdx.x;
  long s = (long)gridDim.x * blockDim.x;
  for (; i < n; i += s) t16[i] = (_Float16)(e1 * h[i] + aggr[i]);
}

__global__ void k_pool(const float* __restrict__ h, const int* __restrict__ batch,
                       float* __restrict__ gpool, long n) {
  long i = (long)blockIdx.x * blockDim.x + threadIdx.x;
  long s = (long)gridDim.x * blockDim.x;
  for (; i < n; i += s) {
    long node = i >> 6;
    int col = (int)(i & 63);
    unsafeAtomicAdd(&gpool[(long)batch[node] * 64 + col], h[i]);
  }
}

// ---------------------------------------------------------------------- launch
static inline size_t align256(size_t x) { return (x + 255) & ~(size_t)255; }

extern "C" void kernel_launch(void* const* d_in, const int* in_sizes, int n_in,
                              void* d_out, int out_size, void* d_ws, size_t ws_size,
                              hipStream_t stream) {
  const float* x         = (const float*)d_in[0];
  const float* edge_attr = (const float*)d_in[1];
  const float* W_in = (const float*)d_in[2];
  const float* b_in = (const float*)d_in[3];
  const float* g_in = (const float*)d_in[4];
  const float* bb_in = (const float*)d_in[5];
  const float* We  = (const float*)d_in[6];
  const float* be  = (const float*)d_in[7];
  const float* ge  = (const float*)d_in[8];
  const float* bbe = (const float*)d_in[9];
  const float* Wc  = (const float*)d_in[10];
  const float* eps = (const float*)d_in[11];
  const float* gn  = (const float*)d_in[12];
  const float* bnb = (const float*)d_in[13];
  const float* Wo1 = (const float*)d_in[14];
  const float* bo1 = (const float*)d_in[15];
  const float* go  = (const float*)d_in[16];
  const float* bbo = (const float*)d_in[17];
  const float* Wo2 = (const float*)d_in[18];
  const float* bo2 = (const float*)d_in[19];
  const int* edge_index = (const int*)d_in[20];
  const int* batch      = (const int*)d_in[21];
  float* out = (float*)d_out;

  const long N = in_sizes[0] / 64;
  const long E = in_sizes[1] / 16;
  const int* src = edge_index;
  const int* dst = edge_index + E;

  // workspace carve-up
  char* ws = (char*)d_ws;
  size_t o = 0;
  auto take = [&](size_t bytes) { size_t r = o; o += align256(bytes); return r; };
  float*    h_    = (float*)   (ws + take((size_t)N * 64 * 4));
  _Float16* h16   = (_Float16*)(ws + take((size_t)N * 64 * 2));
  float*    aggr  = (float*)   (ws + take((size_t)N * 64 * 4));
  float*    Z     = (float*)   (ws + take((size_t)N * 64 * 4));
  _Float16* t16   = (_Float16*)(ws + take((size_t)N * 64 * 2));   // also x16
  _Float16* ea16  = (_Float16*)(ws + take((size_t)E * 16 * 2));
  _Float16* WinT  = (_Float16*)(ws + take(64 * 64 * 2));
  _Float16* WcT   = (_Float16*)(ws + take(3 * 64 * 64 * 2));
  _Float16* WeT   = (_Float16*)(ws + take(3 * 64 * 32 * 2));
  _Float16* Wo1T  = (_Float16*)(ws + take(64 * 64 * 2));
  _Float16* Wo2T  = (_Float16*)(ws + take(64 * 64 * 2));
  float*    stats = (float*)   (ws + take(128 * 4));
  float*    ssb   = (float*)   (ws + take(128 * 4));
  float*    gpool = (float*)   (ws + take((size_t)NG * 64 * 4));
  _Float16* gp16  = (_Float16*)(ws + take((size_t)NG * 64 * 2));
  float*    Zo    = (float*)   (ws + take((size_t)NG * 64 * 4));
  float*    hid   = (float*)   (ws + take((size_t)NG * 64 * 4));
  _Float16* hid16 = (_Float16*)(ws + take((size_t)NG * 64 * 2));
  (void)ws_size; (void)n_in; (void)out_size;

  const long NH = N * 64;
  const int EW = 512;                         // elementwise grid
  const long Mtiles = N / 16, Etiles = E / 16, Gtiles = NG / 16;
  const int nodeGemmBlks = (int)((Mtiles + 7) / 8);
  const int edgeBlks = (int)((Etiles + 7) / 8);
  const int statBlks = 512;

  // ---- one-time prep: f16 conversions + transposed weights
  k_f32_to_f16<<<EW, 256, 0, stream>>>(x, t16, NH);
  k_f32_to_f16<<<EW, 256, 0, stream>>>(edge_attr, ea16, E * 16);
  k_prep_wT<<<16, 256, 0, stream>>>(W_in, WinT, 64, 64);
  for (int l = 0; l < 3; ++l) {
    k_prep_wT<<<16, 256, 0, stream>>>(Wc + (size_t)l * 64 * 64, WcT + (size_t)l * 64 * 64, 64, 64);
    k_prep_wT<<<8, 256, 0, stream>>>(We + (size_t)l * 16 * 64, WeT + (size_t)l * 64 * 32, 16, 32);
  }
  k_prep_wT<<<16, 256, 0, stream>>>(Wo1, Wo1T, 64, 64);
  k_prep_wT<<<16, 256, 0, stream>>>(Wo2, Wo2T, 64, 64);

  // ---- input encoder: h = relu(BN(x @ W_in + b_in))
  k_gemm64<<<nodeGemmBlks, 256, 0, stream>>>(t16, WinT, b_in, Z, Mtiles);
  hipMemsetAsync(stats, 0, 128 * 4, stream);
  k_col_stats<<<statBlks, 256, 0, stream>>>(Z, N, stats);
  k_bn_finalize<<<1, 64, 0, stream>>>(stats, 1.0f / (float)N, g_in, bb_in, ssb);
  k_bn_relu<<<EW, 256, 0, stream>>>(Z, ssb, h_, h16, NH, 0);

  // ---- 3 GINE layers
  for (int l = 0; l < 3; ++l) {
    const _Float16* WeTl = WeT + (size_t)l * 64 * 32;
    const float* bel = be + l * 64;
    // edge BN stats (two-pass, no materialized ea)
    hipMemsetAsync(stats, 0, 128 * 4, stream);
    k_edge_pass1<<<edgeBlks, 256, 0, stream>>>(ea16, WeTl, bel, stats, Etiles);
    k_bn_finalize<<<1, 64, 0, stream>>>(stats, 1.0f / (float)E, ge + l * 64, bbe + l * 64, ssb);
    // message + scatter
    hipMemsetAsync(aggr, 0, (size_t)NH * 4, stream);
    k_edge_pass2<<<edgeBlks, 256, 0, stream>>>(ea16, WeTl, bel, ssb, src, dst, h_, aggr, Etiles);
    // node update
    k_combine<<<EW, 256, 0, stream>>>(h_, aggr, eps, l, t16, NH);
    k_gemm64<<<nodeGemmBlks, 256, 0, stream>>>(t16, WcT + (size_t)l * 64 * 64, nullptr, Z, Mtiles);
    hipMemsetAsync(stats, 0, 128 * 4, stream);
    k_col_stats<<<statBlks, 256, 0, stream>>>(Z, N, stats);
    k_bn_finalize<<<1, 64, 0, stream>>>(stats, 1.0f / (float)N, gn + l * 64, bnb + l * 64, ssb);
    k_bn_relu<<<EW, 256, 0, stream>>>(Z, ssb, h_, h16, NH, 1);  // residual (prev == h)
  }

  // ---- pooling + output MLP
  hipMemsetAsync(gpool, 0, (size_t)NG * 64 * 4, stream);
  k_pool<<<EW, 256, 0, stream>>>(h_, batch, gpool, NH);
  k_f32_to_f16<<<16, 256, 0, stream>>>(gpool, gp16, (long)NG * 64);
  k_gemm64<<<(int)((Gtiles + 7) / 8), 256, 0, stream>>>(gp16, Wo1T, bo1, Zo, Gtiles);
  hipMemsetAsync(stats, 0, 128 * 4, stream);
  k_col_stats<<<32, 256, 0, stream>>>(Zo, NG, stats);
  k_bn_finalize<<<1, 64, 0, stream>>>(stats, 1.0f / (float)NG, go, bbo, ssb);
  k_bn_relu<<<16, 256, 0, stream>>>(Zo, ssb, hid, hid16, (long)NG * 64, 0);
  k_gemm64<<<(int)((Gtiles + 7) / 8), 256, 0, stream>>>(hid16, Wo2T, bo2, out, Gtiles);
}
